// RNN_69303592288921
// MI455X (gfx1250) — compile-verified, matrix operand order checked by reading, not run on hardware
//
#include <hip/hip_runtime.h>

typedef __attribute__((ext_vector_type(16))) _Float16 v16h;
typedef __attribute__((ext_vector_type(8)))  _Float16 v8h;
typedef __attribute__((ext_vector_type(8)))  float    v8f;

#define BB 8
#define SS 256
#define VV 32000
#define HH 1024

// ---- WMMA tile loaders ------------------------------------------------------
// A (16x32, MxK) and B (32x16, KxN) tiles for v_wmma_f32_16x16x32_f16 share the
// same per-lane addressing when the source is row-major along K:
//   elements 0..7 -> K = kb..kb+7, 8..15 -> kb+16..kb+23, kb = kw + 8*(lane>>4)
__device__ __forceinline__ v16h ldA_f16(const _Float16* __restrict__ row, int kb) {
  v8h lo = *(const v8h*)(row + kb);
  v8h hi = *(const v8h*)(row + kb + 16);
  v16h r;
#pragma unroll
  for (int i = 0; i < 8; ++i) { r[i] = lo[i]; r[i + 8] = hi[i]; }
  return r;
}

__device__ __forceinline__ v16h ldA_f32(const float* __restrict__ row, int kb) {
  v8f lo = *(const v8f*)(row + kb);
  v8f hi = *(const v8f*)(row + kb + 16);
  v16h r;
#pragma unroll
  for (int i = 0; i < 8; ++i) { r[i] = (_Float16)lo[i]; r[i + 8] = (_Float16)hi[i]; }
  return r;
}

__device__ __forceinline__ v8f wmma_f16(v16h a, v16h b, v8f c) {
  return __builtin_amdgcn_wmma_f32_16x16x32_f16(false, a, false, b, (short)0, c,
                                                false, false);
}

// ---- Register-blocked NT GEMM: C[64x32 per wave] = A @ B^T (+ cv) -----------
// 4 M-subtiles x 2 N-subtiles of 16x16 per wave: 6 tile loads feed 8 WMMAs per
// K-window (48B/WMMA vs 128B/WMMA unblocked). K = HH fixed. m-fastest supertile
// order -> B (e.g. the 64MB vocab matrix G) stays L2-resident and is reused.
template <bool A_F32, bool OUT_F16>
__global__ void __launch_bounds__(128) gemm_nt_kernel(
    const void* __restrict__ Ap,      // [mSuper*64, HH] f32 or f16
    const _Float16* __restrict__ Bp,  // [nSuper*32, HH] f16
    void* __restrict__ Cp,            // [mSuper*64, ldc] f16 or f32
    const float* __restrict__ cv,     // optional column bias (f32 out path)
    int mSuper, int nSuper, size_t ldc)
{
  const int lane  = threadIdx.x & 31;
  const int wave  = threadIdx.x >> 5;
  const int sid   = blockIdx.x * 4 + wave;
  const int sm    = sid % mSuper;        // m fastest -> consecutive waves share B
  const int sn    = sid / mSuper;
  const int m0    = sm * 64;
  const int n0    = sn * 32;
  const int l15   = lane & 15;
  const int khalf = (lane >> 4) << 3;

  const float*    Af = (const float*)Ap;
  const _Float16* Ah = (const _Float16*)Ap;

  const float*    af[4];
  const _Float16* ah[4];
  const _Float16* bh[2];
#pragma unroll
  for (int i = 0; i < 4; ++i) {
    if (A_F32) af[i] = Af + (size_t)(m0 + i * 16 + l15) * HH;
    else       ah[i] = Ah + (size_t)(m0 + i * 16 + l15) * HH;
  }
#pragma unroll
  for (int j = 0; j < 2; ++j) bh[j] = Bp + (size_t)(n0 + j * 16 + l15) * HH;

  v8f acc[4][2];
#pragma unroll
  for (int i = 0; i < 4; ++i)
#pragma unroll
    for (int j = 0; j < 2; ++j) acc[i][j] = (v8f){};

  for (int kw = 0; kw < HH; kw += 32) {
    const int kb = kw + khalf;
    v16h a[4], b[2];
#pragma unroll
    for (int i = 0; i < 4; ++i) a[i] = A_F32 ? ldA_f32(af[i], kb) : ldA_f16(ah[i], kb);
#pragma unroll
    for (int j = 0; j < 2; ++j) b[j] = ldA_f16(bh[j], kb);
#pragma unroll
    for (int i = 0; i < 4; ++i)
#pragma unroll
      for (int j = 0; j < 2; ++j) acc[i][j] = wmma_f16(a[i], b[j], acc[i][j]);
  }

  const int mb = (lane >> 4) << 3;
#pragma unroll
  for (int j = 0; j < 2; ++j) {
    const int   col = n0 + j * 16 + l15;
    const float add = OUT_F16 ? 0.f : (cv ? cv[col] : 0.f);
#pragma unroll
    for (int i = 0; i < 4; ++i) {
#pragma unroll
      for (int r = 0; r < 8; ++r) {
        const size_t row = (size_t)(m0 + i * 16 + mb + r);
        if (OUT_F16) ((_Float16*)Cp)[row * ldc + col] = (_Float16)acc[i][j][r];
        else         ((float*)Cp)[row * ldc + col] = acc[i][j][r] + add;
      }
    }
  }
}

// ---- RNN step: hOut = tanh(inpA @ Wa^T + hIn @ Wh^T + bias) -----------------
// K split across the 4 waves of a block (each wave: H/4 of K -> 16 WMMAs on the
// critical path instead of 64), partials reduced through LDS, fused bias+tanh.
// Grid = HH/16 = 64 blocks; 256 waves live during the scan.
__global__ void __launch_bounds__(128) rnn_step_kernel(
    const _Float16* __restrict__ inpA, size_t inp_ld, int inp_rows,
    const _Float16* __restrict__ Wa,
    const _Float16* __restrict__ hIn,
    const _Float16* __restrict__ Wh,
    const float* __restrict__ bias,
    _Float16* __restrict__ hOut,
    _Float16* __restrict__ ysOut, size_t ys_ld)
{
  __shared__ float red[4 * 256];
  const int lane  = threadIdx.x & 31;
  const int wave  = threadIdx.x >> 5;    // K-split index 0..3
  const int n0    = blockIdx.x * 16;
  const int l15   = lane & 15;
  const int khalf = (lane >> 4) << 3;

  const int ar = (l15 < inp_rows) ? l15 : (inp_rows - 1);  // clamp padded rows
  const _Float16* aRow  = inpA + (size_t)ar * inp_ld;
  const _Float16* hRow  = hIn  + (size_t)l15 * HH;
  const _Float16* waRow = Wa   + (size_t)(n0 + l15) * HH;
  const _Float16* whRow = Wh   + (size_t)(n0 + l15) * HH;

  v8f acc = {};
  const int k0 = wave * (HH / 4);
  const int k1 = k0 + (HH / 4);
  for (int kw = k0; kw < k1; kw += 32) {
    const int kb = kw + khalf;
    acc = wmma_f16(ldA_f16(aRow, kb), ldA_f16(waRow, kb), acc);
    acc = wmma_f16(ldA_f16(hRow, kb), ldA_f16(whRow, kb), acc);
  }

  const int mb = (lane >> 4) << 3;
#pragma unroll
  for (int r = 0; r < 8; ++r)
    red[wave * 256 + (mb + r) * 16 + l15] = acc[r];
  __syncthreads();

  for (int e = threadIdx.x; e < 256; e += 128) {
    const float s = red[e] + red[256 + e] + red[512 + e] + red[768 + e];
    const int   m = e >> 4;
    const int   n = n0 + (e & 15);
    const _Float16 hv = (_Float16)tanhf(s + bias[n]);
    hOut[(size_t)m * HH + n] = hv;
    if (ysOut != nullptr && m < BB) ysOut[(size_t)m * ys_ld + n] = hv;
  }
}

// ---- row softmax over V -----------------------------------------------------
__global__ void __launch_bounds__(256) softmax_kernel(float* __restrict__ out)
{
  __shared__ float red[8];
  float* p = out + (size_t)blockIdx.x * VV;
  const int tid = threadIdx.x;

  float mx = -3.4e38f;
  for (int i = tid; i < VV; i += 256) mx = fmaxf(mx, p[i]);
  for (int o = 16; o > 0; o >>= 1) mx = fmaxf(mx, __shfl_xor(mx, o));
  if ((tid & 31) == 0) red[tid >> 5] = mx;
  __syncthreads();
  mx = red[0];
  for (int w = 1; w < 8; ++w) mx = fmaxf(mx, red[w]);

  float s = 0.f;
  for (int i = tid; i < VV; i += 256) s += __expf(p[i] - mx);
  for (int o = 16; o > 0; o >>= 1) s += __shfl_xor(s, o);
  __syncthreads();
  if ((tid & 31) == 0) red[tid >> 5] = s;
  __syncthreads();
  s = 0.f;
  for (int w = 0; w < 8; ++w) s += red[w];
  const float inv = 1.0f / s;
  for (int i = tid; i < VV; i += 256) p[i] = __expf(p[i] - mx) * inv;
}

// ---- prep kernels -----------------------------------------------------------
__global__ void cvt_f16_kernel(const float* __restrict__ s, _Float16* __restrict__ d, int n) {
  const int i = blockIdx.x * blockDim.x + threadIdx.x;
  if (i < n) d[i] = (_Float16)s[i];
}

__global__ void transpose_f16_kernel(const float* __restrict__ s, _Float16* __restrict__ d) {
  const int j = blockIdx.x;                      // d[j,h] = s[h,j]
  for (int h = threadIdx.x; h < HH; h += blockDim.x)
    d[(size_t)j * HH + h] = (_Float16)s[(size_t)h * HH + j];
}

__global__ void bias_kernel(const float* bi, const float* bh, const float* __restrict__ Wi1,
                            const float* __restrict__ bo0, float* bias0, float* b1eff) {
  const int n = blockIdx.x * blockDim.x + threadIdx.x;
  if (n >= HH) return;
  bias0[n] = bi[n] + bh[n];
  float s = bi[HH + n] + bh[HH + n];
  for (int j = 0; j < HH; ++j) s += Wi1[(size_t)n * HH + j] * bo0[j];
  b1eff[n] = s;
}

__global__ void cv_kernel(const float* __restrict__ E, const float* __restrict__ bo1,
                          float* __restrict__ cv) {
  const int v = blockIdx.x * blockDim.x + threadIdx.x;
  if (v >= VV) return;
  float s = 0.f;
  for (int h = 0; h < HH; ++h) s += E[(size_t)v * HH + h] * bo1[h];
  cv[v] = s;
}

__global__ void gather_kernel(const int* __restrict__ tok, const float* __restrict__ E,
                              _Float16* __restrict__ x16) {
  const int r = blockIdx.x;
  const int t = tok[r];
  for (int h = threadIdx.x; h < HH; h += blockDim.x)
    x16[(size_t)r * HH + h] = (_Float16)E[(size_t)t * HH + h];
}

__global__ void zero_kernel(_Float16* p, int n) {
  const int i = blockIdx.x * blockDim.x + threadIdx.x;
  if (i < n) p[i] = (_Float16)0.f;
}

// ---- host -------------------------------------------------------------------
extern "C" void kernel_launch(void* const* d_in, const int* in_sizes, int n_in,
                              void* d_out, int out_size, void* d_ws, size_t ws_size,
                              hipStream_t stream) {
  (void)in_sizes; (void)n_in; (void)out_size; (void)ws_size;
  const int*   tokens = (const int*)d_in[0];
  const float* E  = (const float*)d_in[1];
  const float* Wi = (const float*)d_in[2];
  const float* bi = (const float*)d_in[3];
  const float* Wh = (const float*)d_in[4];
  const float* bh = (const float*)d_in[5];
  const float* Wo = (const float*)d_in[6];
  const float* bo = (const float*)d_in[7];
  float* out = (float*)d_out;

  char* w = (char*)d_ws;
  auto carve = [&](size_t bytes) -> char* {
    char* p = w; w += (bytes + 255) & ~(size_t)255; return p;
  };
  _Float16* x16   = (_Float16*)carve((size_t)BB * SS * HH * 2);
  _Float16* ys16  = (_Float16*)carve((size_t)BB * SS * HH * 2);
  _Float16* Wi0h  = (_Float16*)carve((size_t)HH * HH * 2);
  _Float16* Wh0h  = (_Float16*)carve((size_t)HH * HH * 2);
  _Float16* Wh1h  = (_Float16*)carve((size_t)HH * HH * 2);
  _Float16* A1h   = (_Float16*)carve((size_t)HH * HH * 2);
  _Float16* Wo0Th = (_Float16*)carve((size_t)HH * HH * 2);
  _Float16* Wo1Th = (_Float16*)carve((size_t)HH * HH * 2);
  _Float16* Gh    = (_Float16*)carve((size_t)VV * HH * 2);
  _Float16* hb    = (_Float16*)carve((size_t)4 * 16 * HH * 2);
  float* bias0 = (float*)carve((size_t)HH * 4);
  float* b1eff = (float*)carve((size_t)HH * 4);
  float* cv    = (float*)carve((size_t)VV * 4);

  _Float16* h0[2] = { hb,            hb + 16 * HH };
  _Float16* h1[2] = { hb + 32 * HH,  hb + 48 * HH };

  const int HHn = HH * HH;
  cvt_f16_kernel<<<(HHn + 255) / 256, 256, 0, stream>>>(Wi, Wi0h, HHn);
  cvt_f16_kernel<<<(HHn + 255) / 256, 256, 0, stream>>>(Wh, Wh0h, HHn);
  cvt_f16_kernel<<<(HHn + 255) / 256, 256, 0, stream>>>(Wh + (size_t)HHn, Wh1h, HHn);
  transpose_f16_kernel<<<HH, 256, 0, stream>>>(Wo, Wo0Th);
  transpose_f16_kernel<<<HH, 256, 0, stream>>>(Wo + (size_t)HHn, Wo1Th);
  bias_kernel<<<(HH + 255) / 256, 256, 0, stream>>>(bi, bh, Wi + (size_t)HHn, bo, bias0, b1eff);
  cv_kernel<<<(VV + 255) / 256, 256, 0, stream>>>(E, bo + HH, cv);
  gather_kernel<<<BB * SS, 256, 0, stream>>>(tokens, E, x16);
  zero_kernel<<<(4 * 16 * HH + 255) / 256, 256, 0, stream>>>(hb, 4 * 16 * HH);

  // A1 = Wi1 @ Wo0 (WMMA): 16x32 supertiles -> 512 waves / 4 = 128 blocks
  gemm_nt_kernel<true, true><<<(16 * 32) / 4, 128, 0, stream>>>(
      Wi + (size_t)HHn, Wo0Th, A1h, nullptr, 16, 32, (size_t)HH);

  // G = E @ Wo1 (WMMA): 500x32 supertiles -> 4000 blocks
  gemm_nt_kernel<true, true><<<(500 * 32) / 4, 128, 0, stream>>>(
      E, Wo1Th, Gh, nullptr, 500, 32, (size_t)HH);

  // Sequential scan: 2 launches per step (Wo0 folded into A1), K-split waves
  for (int t = 0; t < SS; ++t) {
    rnn_step_kernel<<<HH / 16, 128, 0, stream>>>(
        x16 + (size_t)t * HH, (size_t)SS * HH, BB,
        Wi0h, h0[t & 1], Wh0h, bias0, h0[(t + 1) & 1], (_Float16*)nullptr, (size_t)0);
    rnn_step_kernel<<<HH / 16, 128, 0, stream>>>(
        h0[(t + 1) & 1], (size_t)HH, 16,
        A1h, h1[t & 1], Wh1h, b1eff, h1[(t + 1) & 1],
        ys16 + (size_t)t * HH, (size_t)SS * HH);
  }

  // logits = ys @ G^T + cv : 32x1000 supertiles -> 8000 blocks, then softmax
  gemm_nt_kernel<false, false><<<(32 * 1000) / 4, 128, 0, stream>>>(
      ys16, Gh, out, cv, 32, 1000, (size_t)VV);
  softmax_kernel<<<BB * SS, 256, 0, stream>>>(out);
}
